// SDPAttention_28174985462085
// MI455X (gfx1250) — matrix operand order repaired and checked by reading, not compiled, not run on hardware
//
#include <hip/hip_runtime.h>
#include <hip/hip_bf16.h>
#include <math.h>

// Problem constants (from reference): B=4, S=4096, E=512, H=64
#define B_DIM 4
#define S_DIM 4096
#define E_DIM 512
#define H_DIM 64
#define M_DIM (B_DIM * S_DIM)   // 16384 total rows

typedef _Float16 h16;
typedef __attribute__((ext_vector_type(2)))  _Float16 v2h;
typedef __attribute__((ext_vector_type(4)))  _Float16 v4h;
typedef __attribute__((ext_vector_type(8)))  _Float16 v8h;
typedef __attribute__((ext_vector_type(16))) _Float16 v16h;
typedef __attribute__((ext_vector_type(8)))  float    v8f;

union AF { v16h v; v8h h[2]; unsigned u[8]; };

__device__ __forceinline__ v8f wmma_f16(const v16h& a, const v16h& b, const v8f& c) {
    // D = A(16x32 f16) * B(32x16 f16) + C(16x16 f32)
    return __builtin_amdgcn_wmma_f32_16x16x32_f16(false, a, false, b, (short)0, c,
                                                  false, false);
}

// ---------------------------------------------------------------------------
// Kernel 1: QKV projection.  [M,E] fp32 @ [E,H] fp32 + bias -> f16 workspace.
// blockIdx.y selects (x,W,b,dst).  Q gets 1/sqrt(H)*log2(e) folded in, so the
// attention kernel works directly in the base-2 exponent domain.
// Q,K row-major [M,64] f16; V transposed [B,64,S] f16.
// ---------------------------------------------------------------------------
__global__ __launch_bounds__(128) void qkv_proj_kernel(
    const float* __restrict__ xq, const float* __restrict__ xk,
    const float* __restrict__ xv,
    const float* __restrict__ Wq, const float* __restrict__ bq,
    const float* __restrict__ Wk, const float* __restrict__ bk,
    const float* __restrict__ Wv, const float* __restrict__ bv,
    h16* __restrict__ qws, h16* __restrict__ kws, h16* __restrict__ vws)
{
    const int which = blockIdx.y;
    const float* X;  const float* W;  const float* bias;
    if (which == 0)      { X = xq; W = Wq; bias = bq; }
    else if (which == 1) { X = xk; W = Wk; bias = bk; }
    else                 { X = xv; W = Wv; bias = bv; }

    __shared__ __align__(16) h16 Xs[64][40];   // 64 rows x 32 K (+8 pad)
    __shared__ __align__(16) h16 Ws[64][40];   // W^T tile: [n][kk]

    const int tid  = threadIdx.x;
    const int wave = tid >> 5;
    const int lane = tid & 31;
    const int l15  = lane & 15;
    const int hi   = lane >> 4;
    const int row0 = blockIdx.x * 64;

    v8f acc[4];
    #pragma unroll
    for (int nt = 0; nt < 4; ++nt) {
        float bb = bias[nt * 16 + l15];
        #pragma unroll
        for (int r = 0; r < 8; ++r) acc[nt][r] = bb;
    }

    for (int k0 = 0; k0 < E_DIM; k0 += 32) {
        __syncthreads();
        #pragma unroll
        for (int i = 0; i < 4; ++i) {
            int idx = tid + 128 * i;            // 0..511
            int r   = idx >> 3;                 // 0..63
            int cg  = idx & 7;                  // float4 group
            float4 x4 = *(const float4*)(X + (size_t)(row0 + r) * E_DIM + k0 + cg * 4);
            v4h hv;
            hv.x = (h16)x4.x; hv.y = (h16)x4.y; hv.z = (h16)x4.z; hv.w = (h16)x4.w;
            *(v4h*)&Xs[r][cg * 4] = hv;
        }
        #pragma unroll
        for (int i = 0; i < 4; ++i) {
            int idx = tid + 128 * i;            // 0..511
            int kk  = idx >> 4;                 // 0..31
            int ng  = idx & 15;                 // 0..15
            float4 w4 = *(const float4*)(W + (size_t)(k0 + kk) * H_DIM + ng * 4);
            Ws[ng * 4 + 0][kk] = (h16)w4.x;
            Ws[ng * 4 + 1][kk] = (h16)w4.y;
            Ws[ng * 4 + 2][kk] = (h16)w4.z;
            Ws[ng * 4 + 3][kk] = (h16)w4.w;
        }
        __syncthreads();

        AF a;
        const int ar = wave * 16 + l15;
        a.h[0] = *(const v8h*)&Xs[ar][8 * hi];
        a.h[1] = *(const v8h*)&Xs[ar][16 + 8 * hi];
        #pragma unroll
        for (int nt = 0; nt < 4; ++nt) {
            AF bfr;
            const int bn = nt * 16 + l15;
            bfr.h[0] = *(const v8h*)&Ws[bn][8 * hi];
            bfr.h[1] = *(const v8h*)&Ws[bn][16 + 8 * hi];
            acc[nt] = wmma_f16(a.v, bfr.v, acc[nt]);
        }
    }

    // scale folded into Q: 1/sqrt(64) * log2(e)
    const float oscale = (which == 0) ? 0.18033688011112042f : 1.0f;

    const int m0 = row0 + wave * 16;
    if (which < 2) {
        h16* out = (which == 0) ? qws : kws;     // row-major [M,64]
        #pragma unroll
        for (int nt = 0; nt < 4; ++nt) {
            int col = nt * 16 + l15;
            #pragma unroll
            for (int r = 0; r < 8; ++r) {
                int m = m0 + r + 8 * hi;
                out[(size_t)m * H_DIM + col] = (h16)(acc[nt][r] * oscale);
            }
        }
    } else {
        const int b  = m0 / S_DIM;
        const int s0 = m0 - b * S_DIM;
        #pragma unroll
        for (int nt = 0; nt < 4; ++nt) {
            int col = nt * 16 + l15;
            v8h pk;
            #pragma unroll
            for (int r = 0; r < 8; ++r) pk[r] = (h16)acc[nt][r];
            *(v8h*)&vws[((size_t)b * H_DIM + col) * S_DIM + s0 + 8 * hi] = pk;
        }
    }
}

// ---------------------------------------------------------------------------
// Attention inner block (64 keys), fully transposed formulation:
//   S^T = K * Q^T   (keys per VGPR-row, queries per lane)
//   Z^T = V^T * P^T (h per VGPR-row, queries per lane)
// One softmax update per 64 keys; Z rescale only when the block max actually
// grows (wave-uniform ballot guard, EXEC stays all-ones for WMMA legality).
// P^T B-fragments assembled in registers: pack f32->f16 pairs, exchange the
// half-wave partner via shfl_xor(16), select by half. No LDS in the hot loop.
// ---------------------------------------------------------------------------
template <bool MASK>
__device__ __forceinline__ void attn_block64(
    int kb, const h16* __restrict__ kbase, const h16* __restrict__ vbase,
    const AF& qf0, const AF& qf1, v8f z[4], float& mrun, float& lrun,
    int qb, int l15, int hi)
{
    // ---- scores S^T: four 16-key tiles, k-split over h (0..31, 32..63)
    v8f sc[4];
    #pragma unroll
    for (int kt = 0; kt < 4; ++kt) {
        const v8h* krow = (const v8h*)(kbase + (size_t)(kb + kt * 16 + l15) * H_DIM);
        AF a0, a1;
        a0.h[0] = krow[2 * hi];     a0.h[1] = krow[2 * hi + 1];
        a1.h[0] = krow[4 + 2 * hi]; a1.h[1] = krow[4 + 2 * hi + 1];
        v8f s = {};
        s = wmma_f16(a0.v, qf0.v, s);
        s = wmma_f16(a1.v, qf1.v, s);
        sc[kt] = s;
    }

    const int query = qb + l15;
    if (MASK) {
        #pragma unroll
        for (int kt = 0; kt < 4; ++kt)
            #pragma unroll
            for (int r = 0; r < 8; ++r) {
                const int key = kb + kt * 16 + r + 8 * hi;
                if (key > query) sc[kt][r] = -INFINITY;
            }
    }

    // ---- block max (in-lane over 32 elements + one half-wave merge)
    float mx = sc[0][0];
    #pragma unroll
    for (int kt = 0; kt < 4; ++kt)
        #pragma unroll
        for (int r = 0; r < 8; ++r) mx = fmaxf(mx, sc[kt][r]);
    mx = fmaxf(mx, __shfl_xor(mx, 16, 32));

    // ---- rescale running state only if the max grew anywhere in the wave
    if (__ballot(mx > mrun) != 0ull) {
        const float mnew = fmaxf(mrun, mx);
        const float f    = __builtin_amdgcn_exp2f(mrun - mnew);
        lrun *= f;
        #pragma unroll
        for (int t = 0; t < 4; ++t)
            #pragma unroll
            for (int r = 0; r < 8; ++r) z[t][r] *= f;
        mrun = mnew;
    }

    // ---- probabilities (base-2 domain; scale*log2e folded into Q)
    float sum = 0.0f;
    #pragma unroll
    for (int kt = 0; kt < 4; ++kt)
        #pragma unroll
        for (int r = 0; r < 8; ++r) {
            const float p = __builtin_amdgcn_exp2f(sc[kt][r] - mrun);
            sc[kt][r] = p;
            sum += p;
        }
    sum += __shfl_xor(sum, 16, 32);
    lrun += sum;

    // ---- two 32-key sub-blocks: assemble P^T B-fragment, accumulate Z^T
    #pragma unroll
    for (int sb = 0; sb < 2; ++sb) {
        unsigned own0[4], own1[4], par0[4], par1[4];
        #pragma unroll
        for (int j = 0; j < 4; ++j) {
            v2h p0; p0.x = (h16)sc[2 * sb + 0][2 * j]; p0.y = (h16)sc[2 * sb + 0][2 * j + 1];
            v2h p1; p1.x = (h16)sc[2 * sb + 1][2 * j]; p1.y = (h16)sc[2 * sb + 1][2 * j + 1];
            own0[j] = __builtin_bit_cast(unsigned, p0);
            own1[j] = __builtin_bit_cast(unsigned, p1);
        }
        #pragma unroll
        for (int j = 0; j < 4; ++j) {
            par0[j] = (unsigned)__shfl_xor((int)own0[j], 16, 32);
            par1[j] = (unsigned)__shfl_xor((int)own1[j], 16, 32);
        }
        AF pfrag;
        #pragma unroll
        for (int j = 0; j < 4; ++j) {
            pfrag.u[j]     = hi ? par1[j] : own0[j];   // elements 0..7
            pfrag.u[j + 4] = hi ? own1[j] : par0[j];   // elements 8..15
        }
        #pragma unroll
        for (int t = 0; t < 4; ++t) {
            const v8h* vrow = (const v8h*)(vbase + (size_t)(t * 16 + l15) * S_DIM);
            AF va;
            va.h[0] = vrow[kb / 8 + 4 * sb + 2 * hi];
            va.h[1] = vrow[kb / 8 + 4 * sb + 2 * hi + 1];
            z[t] = wmma_f16(va.v, pfrag.v, z[t]);
        }
    }
}

// ---------------------------------------------------------------------------
// Kernel 2: causal flash attention, transposed accumulation.
// Each wave owns one 16-query tile; mask-free full 64-key blocks + exactly
// one masked 64-key block covering [kfull64, qb+16) (always in-bounds).
// ---------------------------------------------------------------------------
__global__ __launch_bounds__(128) void flash_attn_kernel(
    const h16* __restrict__ qws, const h16* __restrict__ kws,
    const h16* __restrict__ vws, float* __restrict__ out)
{
    const int tid  = threadIdx.x;
    const int wave = tid >> 5;
    const int lane = tid & 31;
    const int l15  = lane & 15;
    const int hi   = lane >> 4;

    const int gt  = blockIdx.x * 4 + wave;            // global query-tile id
    const int tpb = S_DIM / 16;
    const int b   = gt / tpb;
    const int qb  = (gt - b * tpb) * 16;
    const size_t bS = (size_t)b * S_DIM;

    const h16* kbase = kws + bS * H_DIM;
    const h16* vbase = vws + (size_t)b * H_DIM * S_DIM;

    // Q^T B-fragments (resident): lane = query column, K-dim = h contiguous
    AF qf0, qf1;
    {
        const v8h* qrow = (const v8h*)(qws + (bS + qb + l15) * H_DIM);
        qf0.h[0] = qrow[2 * hi];     qf0.h[1] = qrow[2 * hi + 1];      // h 0..31
        qf1.h[0] = qrow[4 + 2 * hi]; qf1.h[1] = qrow[4 + 2 * hi + 1];  // h 32..63
    }

    v8f z[4];
    #pragma unroll
    for (int t = 0; t < 4; ++t)
        #pragma unroll
        for (int r = 0; r < 8; ++r) z[t][r] = 0.0f;
    float mrun = -INFINITY, lrun = 0.0f;

    // full (mask-free) 64-key blocks: kb+63 <= qb
    const int kfull = (qb >= 63) ? ((((qb - 63) >> 6) << 6) + 64) : 0;

    for (int kb = 0; kb < kfull; kb += 64) {
        if (kb + 64 < kfull) {  // prefetch next K/V block (global_prefetch_b8)
            __builtin_prefetch(kbase + (size_t)(kb + 64 + 2 * lane) * H_DIM, 0, 1);
            __builtin_prefetch(vbase + (size_t)(2 * lane) * S_DIM + kb + 64, 0, 1);
        }
        attn_block64<false>(kb, kbase, vbase, qf0, qf1, z, mrun, lrun, qb, l15, hi);
    }
    // one masked 64-key block covers keys [kfull, qb+16); tail keys masked off
    attn_block64<true>(kfull, kbase, vbase, qf0, qf1, z, mrun, lrun, qb, l15, hi);

    // ---- epilogue: Z^T rows are consecutive h per lane -> packed b128 stores
    const float inv = 1.0f / lrun;
    float* obase = out + (bS + qb + l15) * H_DIM;
    #pragma unroll
    for (int t = 0; t < 4; ++t) {
        float4 o0, o1;
        o0.x = z[t][0] * inv; o0.y = z[t][1] * inv;
        o0.z = z[t][2] * inv; o0.w = z[t][3] * inv;
        o1.x = z[t][4] * inv; o1.y = z[t][5] * inv;
        o1.z = z[t][6] * inv; o1.w = z[t][7] * inv;
        *(float4*)(obase + t * 16 + 8 * hi)     = o0;
        *(float4*)(obase + t * 16 + 8 * hi + 4) = o1;
    }
}

// ---------------------------------------------------------------------------
extern "C" void kernel_launch(void* const* d_in, const int* in_sizes, int n_in,
                              void* d_out, int out_size, void* d_ws, size_t ws_size,
                              hipStream_t stream) {
    const float* xq = (const float*)d_in[0];
    const float* xk = (const float*)d_in[1];
    const float* xv = (const float*)d_in[2];
    const float* Wq = (const float*)d_in[3];
    const float* bq = (const float*)d_in[4];
    const float* Wk = (const float*)d_in[5];
    const float* bk = (const float*)d_in[6];
    const float* Wv = (const float*)d_in[7];
    const float* bv = (const float*)d_in[8];
    // d_in[9] = mask: causal, computed analytically in-kernel.

    h16* qws = (h16*)d_ws;                               // [M,64] f16 (pre-scaled)
    h16* kws = qws + (size_t)M_DIM * H_DIM;              // [M,64] f16
    h16* vws = kws + (size_t)M_DIM * H_DIM;              // [B,64,S] f16 (transposed)

    dim3 g1(M_DIM / 64, 3);
    qkv_proj_kernel<<<g1, 128, 0, stream>>>(xq, xk, xv, Wq, bq, Wk, bk, Wv, bv,
                                            qws, kws, vws);

    dim3 g2((B_DIM * S_DIM / 16) / 4);                   // 1024 tiles / 4 waves
    flash_attn_kernel<<<g2, 128, 0, stream>>>(qws, kws, vws, (float*)d_out);
}
